// FrustumProposerSEG_29025388987120
// MI455X (gfx1250) — 8-bit matrix operand order repaired and checked by reading, compiled, not run on hardware
//
#include <hip/hip_runtime.h>
#include <stdint.h>


#define N_FRUST 256
#define NUM_MAGS 6
#define NUM_K 40
#define GRID 25
#define NCELL 625
#define KPAD 640
#define N_PTS 100000
#define IOU_W 0.95f
#define DST_W 0.226f
#define DNS_W 0.05f
#define MAX_DIST 50.0f
#define NMS_THRESH 0.5f
#define DNS_R2 9.0f

typedef __attribute__((ext_vector_type(8))) int v8i;

#define WS_MASK      0
#define WS_INTER     163840
#define WS_DNSCNT    425984
#define WS_AREA      432128
#define WS_KEEP      433152
#define WS_ZC        434176
#define WS_MAGS      435200
#define WS_CENTERS   441344

__global__ void init_kernel(int* __restrict__ dns_cnt) {
    int i = blockIdx.x * blockDim.x + threadIdx.x;
    if (i < N_FRUST * NUM_MAGS) dns_cnt[i] = 0;
}

__global__ __launch_bounds__(128) void geom_kernel(
    const float* __restrict__ frust,
    uint8_t* __restrict__ mask,
    int* __restrict__ area,
    float* __restrict__ zc_out,
    float* __restrict__ mags_out,
    float* __restrict__ centers2_out)
{
    const int n = blockIdx.x;
    const int tid = threadIdx.x;
    const float* c8 = frust + n * 24;

    float bx[4], by[4], bz[4];
#pragma unroll
    for (int p = 0; p < 4; ++p) {
        bx[p] = 0.5f * (c8[(2*p)*3 + 0] + c8[(2*p+1)*3 + 0]);
        by[p] = 0.5f * (c8[(2*p)*3 + 1] + c8[(2*p+1)*3 + 1]);
        bz[p] = 0.5f * (c8[(2*p)*3 + 2] + c8[(2*p+1)*3 + 2]);
    }
    float inv;
    inv = 1.0f / sqrtf(bx[2]*bx[2] + by[2]*by[2]);
    const float lnx = bx[2]*inv, lny = by[2]*inv;
    inv = 1.0f / sqrtf(bx[3]*bx[3] + by[3]*by[3]);
    const float rnx = bx[3]*inv, rny = by[3]*inv;
    const float cx0 = 0.5f*(bx[2]+bx[3]), cy0 = 0.5f*(by[2]+by[3]);
    inv = 1.0f / sqrtf(cx0*cx0 + cy0*cy0);
    const float cnx = cx0*inv, cny = cy0*inv;

    float mn = 1e30f, mx = -1e30f;
#pragma unroll
    for (int p = 0; p < 4; ++p) {
        float m3 = sqrtf(bx[p]*bx[p] + by[p]*by[p] + bz[p]*bz[p]);
        mn = fminf(mn, m3);
        mx = fmaxf(mx, m3);
    }

    int cnt = 0;
    for (int k = tid; k < KPAD; k += blockDim.x) {
        uint8_t v = 0;
        if (k < NCELL) {
            const int ix = k / GRID, iy = k % GRID;
            const float px = -54.0f + (float)ix * 4.5f;
            const float py = -54.0f + (float)iy * 4.5f;
            const float mag = sqrtf(px*px + py*py);
            const float bnx = px / mag, bny = py / mag;
            float l2x = bnx - lnx, l2y = bny - lny;
            float i1 = 1.0f / sqrtf(l2x*l2x + l2y*l2y);
            l2x *= i1; l2y *= i1;
            float r2x = bnx - rnx, r2y = bny - rny;
            float i2 = 1.0f / sqrtf(r2x*r2x + r2y*r2y);
            r2x *= i2; r2y *= i2;
            const float cos_lr = l2x*r2x + l2y*r2y;
            const float cos_c  = bnx*cnx + bny*cny;
            v = (cos_lr < 0.0f) && (cos_c > 0.0f) && (mag >= mn) && (mag < mx);
        }
        mask[n * KPAD + k] = v;
        cnt += v;
    }

    __shared__ int red[128];
    red[tid] = cnt;
    __syncthreads();
#pragma unroll
    for (int s = 64; s > 0; s >>= 1) {
        if (tid < s) red[tid] += red[tid + s];
        __syncthreads();
    }
    if (tid == 0) {
        area[n] = red[0];
        float zc = 0.0f;
#pragma unroll
        for (int p = 0; p < 8; ++p) zc += c8[p*3 + 2];
        zc_out[n] = zc * 0.125f;
    }
    if (tid < NUM_MAGS) {
        const float t = (float)tid * (1.0f / 5.0f);
        const float mag = mn + (mx - mn) * t;
        mags_out[n * NUM_MAGS + tid] = mag;
        centers2_out[(n * NUM_MAGS + tid) * 2 + 0] = cnx * mag;
        centers2_out[(n * NUM_MAGS + tid) * 2 + 1] = cny * mag;
    }
}

#define PT_TILE 1024
__global__ __launch_bounds__(256) void dns_kernel(
    const float* __restrict__ points,
    const float* __restrict__ centers2,
    int* __restrict__ dns_cnt)
{
    __shared__ float spx[PT_TILE];
    __shared__ float spy[PT_TILE];
    const int base = blockIdx.x * PT_TILE;
    const int npt = min(PT_TILE, N_PTS - base);
    for (int i = threadIdx.x; i < npt; i += blockDim.x) {
        spx[i] = points[(size_t)(base + i) * 3 + 0];
        spy[i] = points[(size_t)(base + i) * 3 + 1];
    }
    __syncthreads();

    float ccx[6], ccy[6];
    int cc[6];
#pragma unroll
    for (int j = 0; j < 6; ++j) {
        const int c = threadIdx.x + j * 256;
        ccx[j] = centers2[c * 2 + 0];
        ccy[j] = centers2[c * 2 + 1];
        cc[j] = 0;
    }
    for (int i = 0; i < npt; ++i) {
        const float px = spx[i], py = spy[i];
#pragma unroll
        for (int j = 0; j < 6; ++j) {
            const float dx = px - ccx[j];
            const float dy = py - ccy[j];
            cc[j] += (dx*dx + dy*dy < DNS_R2) ? 1 : 0;
        }
    }
#pragma unroll
    for (int j = 0; j < 6; ++j)
        atomicAdd(&dns_cnt[threadIdx.x + j * 256], cc[j]);
}

__global__ __launch_bounds__(32) void inter_wmma_kernel(
    const uint8_t* __restrict__ mask,
    int* __restrict__ inter)
{
    const int ti = blockIdx.x & 15;
    const int tj = blockIdx.x >> 4;
    const int lane = threadIdx.x;
    const int l = lane & 15;
    const int hi = lane >> 4;

    const uint8_t* arow = mask + (size_t)(ti * 16 + l) * KPAD;
    const uint8_t* brow = mask + (size_t)(tj * 16 + l) * KPAD;

    v8i acc = {};
#pragma unroll
    for (int k0 = 0; k0 < KPAD; k0 += 64) {
        v8i a, b;
#pragma unroll
        for (int j = 0; j < 8; ++j) {
            const int aoff = k0 + hi * 8  + (j >> 1) * 16 + (j & 1) * 4;
            const int boff = k0 + hi * 8  + (j >> 1) * 16 + (j & 1) * 4;
            a[j] = *(const int*)(arow + aoff);
            b[j] = *(const int*)(brow + boff);
        }
        acc = __builtin_amdgcn_wmma_i32_16x16x64_iu8(false, a, false, b, acc,
                                                     false, false);
    }
#pragma unroll
    for (int r = 0; r < 8; ++r) {
        const int M = r + hi * 8;
        inter[(size_t)(ti * 16 + M) * N_FRUST + (tj * 16 + l)] = acc[r];
    }
}

__global__ __launch_bounds__(256) void nms_kernel(
    const float* __restrict__ scores,
    const int* __restrict__ inter,
    const int* __restrict__ area,
    int* __restrict__ keep_out)
{
    __shared__ float s_sc[N_FRUST];
    __shared__ int s_order[N_FRUST];
    __shared__ int s_keep[N_FRUST];
    const int tid = threadIdx.x;

    s_sc[tid] = scores[tid];
    s_keep[tid] = 1;
    __syncthreads();

    const float si = s_sc[tid];
    int r = 0;
    for (int j = 0; j < N_FRUST; ++j) {
        const float sj = s_sc[j];
        r += (sj > si) || (sj == si && j < tid);
    }
    s_order[r] = tid;
    __syncthreads();

    const int b = s_order[tid];
    const float area_b = (float)area[b];

    for (int it = 0; it < N_FRUST; ++it) {
        const bool active = (s_keep[it] != 0);
        const int a = s_order[it];
        if (active && tid > it) {
            const float ia = (float)inter[(size_t)a * N_FRUST + b];
            const float uni = (float)area[a] + area_b - ia;
            const float iou = ia / fmaxf(uni, 1.0f);
            if (iou > NMS_THRESH) s_keep[tid] = 0;
        }
        __syncthreads();
    }
    keep_out[b] = s_keep[tid];
}

__global__ __launch_bounds__(256) void output_kernel(
    const float* __restrict__ scores,
    const int* __restrict__ labels,
    const float* __restrict__ base_boxes,
    const float* __restrict__ centers2,
    const float* __restrict__ zc,
    const float* __restrict__ mags,
    const int* __restrict__ dns_cnt,
    const int* __restrict__ keep,
    float* __restrict__ out)
{
    const int idx = blockIdx.x * blockDim.x + threadIdx.x;
    if (idx >= N_FRUST * NUM_MAGS * NUM_K) return;
    const int n = idx / (NUM_MAGS * NUM_K);
    const int rem = idx % (NUM_MAGS * NUM_K);
    const int m = rem / NUM_K;
    const int k = rem % NUM_K;

    const int nm = n * NUM_MAGS + m;
    const float cx = centers2[nm * 2 + 0];
    const float cy = centers2[nm * 2 + 1];
    const float z = zc[n];
    const float* bb = base_boxes + ((size_t)labels[n] * NUM_K + k) * 7;

    const float dns = (float)dns_cnt[nm] * (1.0f / (float)N_PTS);
    float sc = scores[n] * IOU_W + dns * DNS_W - mags[nm] * (DST_W / MAX_DIST);
    sc *= (keep[n] != 0) ? 1.0f : 0.0f;

    float* o = out + (size_t)idx * 8;
    o[0] = cx; o[1] = cy; o[2] = z;
    o[3] = bb[3]; o[4] = bb[4]; o[5] = bb[5]; o[6] = bb[6];
    o[7] = sc;
}

extern "C" void kernel_launch(void* const* d_in, const int* in_sizes, int n_in,
                              void* d_out, int out_size, void* d_ws, size_t ws_size,
                              hipStream_t stream) {
    const float* points      = (const float*)d_in[0];
    const float* frust_boxes = (const float*)d_in[1];
    const float* scores      = (const float*)d_in[2];
    const int*   labels      = (const int*)  d_in[3];
    const float* base_boxes  = (const float*)d_in[4];
    float* out = (float*)d_out;

    char* ws = (char*)d_ws;
    uint8_t* mask    = (uint8_t*)(ws + WS_MASK);
    int*     inter   = (int*)    (ws + WS_INTER);
    int*     dns_cnt = (int*)    (ws + WS_DNSCNT);
    int*     area    = (int*)    (ws + WS_AREA);
    int*     keep    = (int*)    (ws + WS_KEEP);
    float*   zc      = (float*)  (ws + WS_ZC);
    float*   mags    = (float*)  (ws + WS_MAGS);
    float*   cent2   = (float*)  (ws + WS_CENTERS);

    init_kernel<<<(N_FRUST * NUM_MAGS + 255) / 256, 256, 0, stream>>>(dns_cnt);

    geom_kernel<<<N_FRUST, 128, 0, stream>>>(frust_boxes, mask, area, zc, mags,
                                             cent2);

    dns_kernel<<<(N_PTS + PT_TILE - 1) / PT_TILE, 256, 0, stream>>>(points,
                                                                    cent2,
                                                                    dns_cnt);

    inter_wmma_kernel<<<(N_FRUST / 16) * (N_FRUST / 16), 32, 0, stream>>>(mask,
                                                                          inter);

    nms_kernel<<<1, N_FRUST, 0, stream>>>(scores, inter, area, keep);

    const int total = N_FRUST * NUM_MAGS * NUM_K;
    output_kernel<<<(total + 255) / 256, 256, 0, stream>>>(
        scores, labels, base_boxes, cent2, zc, mags, dns_cnt, keep, out);
}
